// TransformerEncoder_25340307046997
// MI455X (gfx1250) — matrix operand order repaired and checked by reading, not compiled, not run on hardware
//
#include <hip/hip_runtime.h>
#include <hip/hip_bf16.h>

#define B_ 4
#define S_ 1024
#define F_ 1024
#define H_ 16
#define D_ 64
#define K_DIM F_

typedef __bf16 bf16_t;
typedef __bf16 v16bf __attribute__((ext_vector_type(16)));
typedef float  v8f   __attribute__((ext_vector_type(8)));
typedef unsigned int v4u __attribute__((ext_vector_type(4)));
typedef int v4i __attribute__((ext_vector_type(4)));

__device__ __forceinline__ bf16_t f2bf(float v) { return (bf16_t)v; }

// --------------------------------------------------------------------------
// Optional gfx1250 async global->LDS copy (ASYNCcnt) with sync fallback.
// ROCm 7.2 signature (from probe diagnostics): (v4i as1*, v4i as3*, imm, imm)
// --------------------------------------------------------------------------
#if defined(__has_builtin)
#if __has_builtin(__builtin_amdgcn_global_load_async_to_lds_b128) && \
    __has_builtin(__builtin_amdgcn_s_wait_asynccnt)
#define USE_ASYNC_LDS 1
#endif
#endif

#if defined(USE_ASYNC_LDS)
typedef __attribute__((address_space(1))) v4i gv4i_t;
typedef __attribute__((address_space(3))) v4i lv4i_t;
__device__ __forceinline__ void copy16_g2l(const bf16_t* g, bf16_t* l) {
  __builtin_amdgcn_global_load_async_to_lds_b128((gv4i_t*)(void*)g,
                                                 (lv4i_t*)(void*)l, 0, 0);
}
__device__ __forceinline__ void copy16_wait() {
  __builtin_amdgcn_s_wait_asynccnt(0);
}
#else
__device__ __forceinline__ void copy16_g2l(const bf16_t* g, bf16_t* l) {
  *(v4u*)l = *(const v4u*)g;
}
__device__ __forceinline__ void copy16_wait() {}
#endif

// ---------------------------------------------------------------------------
// WMMA fragment helpers (CDNA5 16x16x32 bf16, wave32) — all loads are 16B.
// A 16x32: lane m=lane&15, g=lane>>4; elem e -> K=(e<8?0:16)+8g+(e&7)
//   => two contiguous 16B runs at K=8g and K=16+8g.
// B 32x16: lane n=lane&15, g; elem e -> K=(e<8?0:16)+8g+(e&7)
//   => two contiguous 16B runs when the source is n-major (src[n*ld+k]).
// C 16x16: elem r -> M=r+8g, N=lane&15
// ---------------------------------------------------------------------------
union Frag16 {
  v16bf v;
  v4u q[2];
};

__device__ __forceinline__ v16bf load_a_frag(const bf16_t* src, int ld) {
  int lane = threadIdx.x & 31;
  const bf16_t* p = src + (size_t)(lane & 15) * ld + 8 * (lane >> 4);
  Frag16 f;
  f.q[0] = *(const v4u*)p;
  f.q[1] = *(const v4u*)(p + 16);
  return f.v;
}

// source is n-major: element value = src[n*ld + k]
__device__ __forceinline__ v16bf load_bn_frag(const bf16_t* src, int ld) {
  int lane = threadIdx.x & 31;
  const bf16_t* p = src + (size_t)(lane & 15) * ld + 8 * (lane >> 4);
  Frag16 f;
  f.q[0] = *(const v4u*)p;
  f.q[1] = *(const v4u*)(p + 16);
  return f.v;
}

__device__ __forceinline__ v8f wmma_bf16(v16bf a, v16bf b, v8f c) {
  return __builtin_amdgcn_wmma_f32_16x16x32_bf16(false, a, false, b, (short)0, c,
                                                 false, false);
}

__device__ __forceinline__ void zero8(v8f& v) {
#pragma unroll
  for (int i = 0; i < 8; ++i) v[i] = 0.0f;
}

// ---------------------------------------------------------------------------
// Conversion / packing kernels (one-time; weights stored TRANSPOSED = n-major)
// ---------------------------------------------------------------------------
__global__ __launch_bounds__(256) void f32_to_bf16_kernel(const float* __restrict__ src,
                                                          bf16_t* __restrict__ dst, int n) {
  int i = blockIdx.x * 256 + threadIdx.x;
  if (i < n) dst[i] = f2bf(src[i]);
}

// W [K][N] fp32 -> Wt [N][K] bf16
__global__ __launch_bounds__(256) void transpose_to_bf16_kernel(const float* __restrict__ src,
                                                                bf16_t* __restrict__ dst) {
  int idx = blockIdx.x * 256 + threadIdx.x;  // idx = n*K + k
  if (idx >= F_ * F_) return;
  int n = idx / K_DIM;
  int k = idx - n * K_DIM;
  dst[idx] = f2bf(src[(size_t)k * F_ + n]);
}

// Wk/Wq/Wv [H][F][D] -> wqkv_t [3F][F] bf16, row c = p*F + h*D + d, col f
__global__ __launch_bounds__(256) void pack_wqkv_kernel(const float* __restrict__ Wk,
                                                        const float* __restrict__ Wq,
                                                        const float* __restrict__ Wv,
                                                        bf16_t* __restrict__ dst) {
  int idx = blockIdx.x * 256 + threadIdx.x;  // idx = c*F + f
  if (idx >= 3 * F_ * F_) return;
  int c = idx / F_;
  int f = idx - c * F_;
  int p = c / F_;
  int rem = c - p * F_;
  int h = rem / D_;
  int d = rem - h * D_;
  const float* W = (p == 0) ? Wk : (p == 1) ? Wq : Wv;
  dst[idx] = f2bf(W[((size_t)h * F_ + f) * D_ + d]);
}

// ---------------------------------------------------------------------------
// QKV GEMM: [4096 x 1024] x [1024 x 3072] with n-major weights; no LDS —
// all operands are L2-resident (192MB), fragments are direct 16B loads.
// Block = 4 waves; wave owns 16 rows x 64 cols. V is stored TRANSPOSED
// ([b][h][d][s]) so the attention PV B-fragments are contiguous.
// ---------------------------------------------------------------------------
__global__ __launch_bounds__(128) void gemm_qkv_kernel(const bf16_t* __restrict__ A,
                                                       const bf16_t* __restrict__ Wt,
                                                       const float* __restrict__ bk,
                                                       const float* __restrict__ bq,
                                                       const float* __restrict__ bv,
                                                       bf16_t* __restrict__ kd,
                                                       bf16_t* __restrict__ qd,
                                                       bf16_t* __restrict__ vd) {
  const int wave = threadIdx.x >> 5;
  const int lane = threadIdx.x & 31;
  const int g = lane >> 4;
  const int n = lane & 15;
  const int m0 = blockIdx.x * 64 + wave * 16;
  const int n0 = blockIdx.y * 64;

  const bf16_t* Arow = A + (size_t)m0 * K_DIM;
  v8f acc[4];
#pragma unroll
  for (int t = 0; t < 4; ++t) zero8(acc[t]);

#pragma unroll 2
  for (int k0 = 0; k0 < K_DIM; k0 += 32) {
    v16bf a = load_a_frag(Arow + k0, K_DIM);
#pragma unroll
    for (int t = 0; t < 4; ++t)
      acc[t] = wmma_bf16(a, load_bn_frag(Wt + (size_t)(n0 + t * 16) * K_DIM + k0, K_DIM),
                         acc[t]);
  }

  const int p = n0 / F_;                // projection (0=k,1=q,2=v), uniform
  const int h = (n0 - p * F_) / D_;     // head, uniform
  const int b = m0 / S_;                // batch, uniform
  const float* bias = (p == 0) ? bk : (p == 1) ? bq : bv;
#pragma unroll
  for (int t = 0; t < 4; ++t) {
#pragma unroll
    for (int r = 0; r < 8; ++r) {
      int s = m0 + r + 8 * g - b * S_;
      int d = t * 16 + n;  // n0 % 64 == 0
      float v = acc[t][r] + bias[h * D_ + d];
      if (p == 0)
        kd[(((size_t)b * H_ + h) * S_ + s) * D_ + d] = f2bf(v);
      else if (p == 1)
        qd[(((size_t)b * H_ + h) * S_ + s) * D_ + d] = f2bf(v);
      else
        vd[(((size_t)b * H_ + h) * D_ + d) * S_ + s] = f2bf(v);  // transposed
    }
  }
}

// ---------------------------------------------------------------------------
// Flash attention: scores = K@Q^T/8 (softmax over q index j), O = P@V.
// Block = (b, h, 64 i-rows); 4 waves, one 16-row tile each. Q[j][d] and
// V^T[d][j] tiles staged in LDS (async copy on gfx1250), P staged per-wave.
// Scores never touch global memory.
// ---------------------------------------------------------------------------
__global__ __launch_bounds__(128) void attn_kernel(const bf16_t* __restrict__ kb,
                                                   const bf16_t* __restrict__ qb,
                                                   const bf16_t* __restrict__ vtb,
                                                   float* __restrict__ att) {
  __shared__ bf16_t qt[32][64];      // [j][d]
  __shared__ bf16_t vt[64][32];      // [d][j]  (V already transposed in memory)
  __shared__ bf16_t pt[4][16][32];   // per-wave P staging (C-layout -> A-layout)

  const int wave = threadIdx.x >> 5;
  const int lane = threadIdx.x & 31;
  const int g = lane >> 4;
  const int n = lane & 15;

  const int itile = blockIdx.x & 15;
  const int h = (blockIdx.x >> 4) & (H_ - 1);
  const int b = blockIdx.x >> 8;
  const int i0 = itile * 64 + wave * 16;

  const size_t headoff = ((size_t)(b * H_ + h)) * S_ * D_;
  const bf16_t* kptr = kb + headoff + (size_t)i0 * D_;

  // K-row A-fragments are invariant over the whole j loop
  v16bf ak0 = load_a_frag(kptr, D_);
  v16bf ak1 = load_a_frag(kptr + 32, D_);

  v8f o[4];
#pragma unroll
  for (int t = 0; t < 4; ++t) zero8(o[t]);
  float m_r[8], l_r[8];
#pragma unroll
  for (int r = 0; r < 8; ++r) { m_r[r] = -1e30f; l_r[r] = 0.0f; }

  for (int j0 = 0; j0 < S_; j0 += 32) {
    __syncthreads();
    // qt: 32 rows x 8 chunks = 256; vt: 64 rows x 4 chunks = 256
    for (int i = threadIdx.x; i < 512; i += 128) {
      if (i < 256) {
        int r = i >> 3, c = (i & 7) * 8;
        copy16_g2l(qb + headoff + (size_t)(j0 + r) * D_ + c, &qt[r][c]);
      } else {
        int idx = i - 256;
        int r = idx >> 2, c = (idx & 3) * 8;
        copy16_g2l(vtb + headoff + (size_t)r * S_ + j0 + c, &vt[r][c]);
      }
    }
    copy16_wait();
    __syncthreads();

    // scores: two 16x16 j-subtiles; B source is n-major (qt[j][d])
    v8f s0, s1;
    zero8(s0);
    zero8(s1);
    s0 = wmma_bf16(ak0, load_bn_frag(&qt[0][0], 64), s0);
    s0 = wmma_bf16(ak1, load_bn_frag(&qt[0][32], 64), s0);
    s1 = wmma_bf16(ak0, load_bn_frag(&qt[16][0], 64), s1);
    s1 = wmma_bf16(ak1, load_bn_frag(&qt[16][32], 64), s1);
#pragma unroll
    for (int r = 0; r < 8; ++r) {
      s0[r] *= 0.125f;  // 1/sqrt(64)
      s1[r] *= 0.125f;
    }

    // online softmax; row stats reduced within the 16-lane half holding the row
#pragma unroll
    for (int r = 0; r < 8; ++r) {
      float mx = fmaxf(s0[r], s1[r]);
      mx = fmaxf(mx, __shfl_xor(mx, 1, 32));
      mx = fmaxf(mx, __shfl_xor(mx, 2, 32));
      mx = fmaxf(mx, __shfl_xor(mx, 4, 32));
      mx = fmaxf(mx, __shfl_xor(mx, 8, 32));
      float nm = fmaxf(m_r[r], mx);
      float sc = __expf(m_r[r] - nm);
      m_r[r] = nm;
      float p0 = __expf(s0[r] - nm);
      float p1 = __expf(s1[r] - nm);
      s0[r] = p0;
      s1[r] = p1;
      float rs = p0 + p1;
      rs += __shfl_xor(rs, 1, 32);
      rs += __shfl_xor(rs, 2, 32);
      rs += __shfl_xor(rs, 4, 32);
      rs += __shfl_xor(rs, 8, 32);
      l_r[r] = l_r[r] * sc + rs;
#pragma unroll
      for (int t = 0; t < 4; ++t) o[t][r] *= sc;
    }

    // C-layout P -> LDS -> A-layout (DS ops are in-order within a wave)
#pragma unroll
    for (int r = 0; r < 8; ++r) {
      pt[wave][r + 8 * g][n] = f2bf(s0[r]);
      pt[wave][r + 8 * g][16 + n] = f2bf(s1[r]);
    }
    v16bf ap = load_a_frag(&pt[wave][0][0], 32);
    // PV: B source is n-major (vt[d][j])
#pragma unroll
    for (int t = 0; t < 4; ++t)
      o[t] = wmma_bf16(ap, load_bn_frag(&vt[t * 16][0], 32), o[t]);
  }

  // normalize rows, scatter heads back to [b][s][h*D+d]
#pragma unroll
  for (int t = 0; t < 4; ++t) {
#pragma unroll
    for (int r = 0; r < 8; ++r) {
      int srow = i0 + r + 8 * g;
      att[((size_t)b * S_ + srow) * F_ + h * D_ + t * 16 + n] = o[t][r] / l_r[r];
    }
  }
}

// ---------------------------------------------------------------------------
// FF GEMM: [4096 x 1024] x W^T [1024 x 1024] (n-major), fused bias/ReLU/residual
// ---------------------------------------------------------------------------
template <bool RELU>
__global__ __launch_bounds__(128) void gemm_ff_kernel(const bf16_t* __restrict__ A,
                                                      const bf16_t* __restrict__ Wt,
                                                      const float* __restrict__ bias,
                                                      const float* __restrict__ res,
                                                      float* __restrict__ outf,
                                                      bf16_t* __restrict__ outb) {
  const int wave = threadIdx.x >> 5;
  const int lane = threadIdx.x & 31;
  const int g = lane >> 4;
  const int n = lane & 15;
  const int m0 = blockIdx.x * 64 + wave * 16;
  const int n0 = blockIdx.y * 64;

  const bf16_t* Arow = A + (size_t)m0 * K_DIM;
  v8f acc[4];
#pragma unroll
  for (int t = 0; t < 4; ++t) zero8(acc[t]);

#pragma unroll 2
  for (int k0 = 0; k0 < K_DIM; k0 += 32) {
    v16bf a = load_a_frag(Arow + k0, K_DIM);
#pragma unroll
    for (int t = 0; t < 4; ++t)
      acc[t] = wmma_bf16(a, load_bn_frag(Wt + (size_t)(n0 + t * 16) * K_DIM + k0, K_DIM),
                         acc[t]);
  }

#pragma unroll
  for (int t = 0; t < 4; ++t) {
#pragma unroll
    for (int r = 0; r < 8; ++r) {
      int row = m0 + r + 8 * g;
      int col = n0 + t * 16 + n;
      float v = acc[t][r] + bias[col];
      if (RELU) v = fmaxf(v, 0.0f);
      if (res) v += res[(size_t)row * F_ + col];
      if (outf) outf[(size_t)row * F_ + col] = v;
      if (outb) outb[(size_t)row * F_ + col] = f2bf(v);
    }
  }
}

// ---------------------------------------------------------------------------
// LayerNorm over (S,F) per batch: two-stage deterministic tree reduction.
// ---------------------------------------------------------------------------
__global__ __launch_bounds__(256) void add_reduce_kernel(const float* __restrict__ x,
                                                         const float* __restrict__ addv,
                                                         float* __restrict__ y,
                                                         float* __restrict__ partials) {
  const int CH = S_ * F_ / 64;
  const int b = blockIdx.x >> 6;
  const int chunk = blockIdx.x & 63;
  const size_t base = (size_t)b * S_ * F_ + (size_t)chunk * CH;
  float s = 0.0f, sq = 0.0f;
  for (int i = threadIdx.x; i < CH; i += 256) {
    float v = addv[base + i];
    if (x) v += x[base + i];
    y[base + i] = v;
    s += v;
    sq += v * v;
  }
  __shared__ float shs[256], shq[256];
  shs[threadIdx.x] = s;
  shq[threadIdx.x] = sq;
  __syncthreads();
  for (int off = 128; off > 0; off >>= 1) {
    if (threadIdx.x < off) {
      shs[threadIdx.x] += shs[threadIdx.x + off];
      shq[threadIdx.x] += shq[threadIdx.x + off];
    }
    __syncthreads();
  }
  if (threadIdx.x == 0) {
    partials[blockIdx.x * 2] = shs[0];
    partials[blockIdx.x * 2 + 1] = shq[0];
  }
}

__global__ __launch_bounds__(64) void finalize_stats_kernel(const float* __restrict__ partials,
                                                            float* __restrict__ stats) {
  __shared__ float shs[64], shq[64];
  const int b = blockIdx.x;
  const int t = threadIdx.x;
  shs[t] = partials[(b * 64 + t) * 2];
  shq[t] = partials[(b * 64 + t) * 2 + 1];
  __syncthreads();
  for (int off = 32; off > 0; off >>= 1) {
    if (t < off) {
      shs[t] += shs[t + off];
      shq[t] += shq[t + off];
    }
    __syncthreads();
  }
  if (t == 0) {
    const float invN = 1.0f / (float)(S_ * F_);
    float mean = shs[0] * invN;
    float var = shq[0] * invN - mean * mean;
    stats[b * 2] = mean;
    stats[b * 2 + 1] = rsqrtf(var);  // reference uses no eps
  }
}

__global__ __launch_bounds__(256) void apply_ln_kernel(const float* __restrict__ y,
                                                       const float* __restrict__ stats,
                                                       const float* __restrict__ gain,
                                                       const float* __restrict__ bias,
                                                       float* __restrict__ outf,
                                                       bf16_t* __restrict__ outb) {
  size_t idx = (size_t)blockIdx.x * 256 + threadIdx.x;
  int b = (int)(idx / (S_ * F_));
  int f = (int)(idx % F_);
  float mean = stats[b * 2];
  float rstd = stats[b * 2 + 1];
  float v = gain[f] * ((y[idx] - mean) * rstd) + bias[f];
  if (outf) outf[idx] = v;
  if (outb) outb[idx] = f2bf(v);
}

// ---------------------------------------------------------------------------
// Launch
// ---------------------------------------------------------------------------
extern "C" void kernel_launch(void* const* d_in, const int* in_sizes, int n_in,
                              void* d_out, int out_size, void* d_ws, size_t ws_size,
                              hipStream_t stream) {
  const float* x   = (const float*)d_in[0];
  const float* Wk  = (const float*)d_in[1];
  const float* bk  = (const float*)d_in[2];
  const float* Wq  = (const float*)d_in[3];
  const float* bq  = (const float*)d_in[4];
  const float* Wv  = (const float*)d_in[5];
  const float* bv  = (const float*)d_in[6];
  const float* W1  = (const float*)d_in[7];
  const float* b1  = (const float*)d_in[8];
  const float* W2  = (const float*)d_in[9];
  const float* b2  = (const float*)d_in[10];
  const float* g1  = (const float*)d_in[11];
  const float* be1 = (const float*)d_in[12];
  const float* g2  = (const float*)d_in[13];
  const float* be2 = (const float*)d_in[14];
  float* out = (float*)d_out;

  char* ws = (char*)d_ws;
  const size_t MB = 1ull << 20;
  bf16_t* x_bf   = (bf16_t*)(ws + 0 * MB);   // 8 MB; reused later as h_bf
  bf16_t* wqkv_t = (bf16_t*)(ws + 8 * MB);   // 6 MB  [3F][F] n-major
  bf16_t* w1_t   = (bf16_t*)(ws + 14 * MB);  // 2 MB  [F][F] n-major
  bf16_t* w2_t   = (bf16_t*)(ws + 16 * MB);  // 2 MB  [F][F] n-major
  bf16_t* k_bf   = (bf16_t*)(ws + 18 * MB);  // 8 MB; reused later as res_bf
  bf16_t* q_bf   = (bf16_t*)(ws + 26 * MB);  // 8 MB ┐ reused later as y2 (16 MB)
  bf16_t* vt_bf  = (bf16_t*)(ws + 34 * MB);  // 8 MB ┘ ([b][h][d][s] transposed)
  float*  att    = (float*)(ws + 42 * MB);   // 16 MB (also y1 = x + att in place)
  float*  resf   = (float*)(ws + 58 * MB);   // 16 MB
  float*  partials = (float*)(ws + 74 * MB); // 2 KB
  float*  stats  = (float*)(ws + 74 * MB + 4096);
  bf16_t* h_bf   = x_bf;            // x_bf dead after QKV GEMM
  bf16_t* res_bf = k_bf;            // k_bf dead after attention
  float*  y2     = (float*)q_bf;    // q/v dead after attention (16 MB contiguous)

  const int NT = B_ * S_ * F_;

  // --- precision staging (fp32 -> bf16; weights transposed to n-major) ---
  f32_to_bf16_kernel<<<NT / 256, 256, 0, stream>>>(x, x_bf, NT);
  pack_wqkv_kernel<<<(3 * F_ * F_) / 256, 256, 0, stream>>>(Wk, Wq, Wv, wqkv_t);
  transpose_to_bf16_kernel<<<(F_ * F_) / 256, 256, 0, stream>>>(W1, w1_t);
  transpose_to_bf16_kernel<<<(F_ * F_) / 256, 256, 0, stream>>>(W2, w2_t);

  // --- QKV projection (V written transposed for the PV GEMM) ---
  gemm_qkv_kernel<<<dim3(64, 48), 128, 0, stream>>>(x_bf, wqkv_t, bk, bq, bv,
                                                    k_bf, q_bf, vt_bf);

  // --- flash attention ---
  attn_kernel<<<B_ * H_ * (S_ / 64), 128, 0, stream>>>(k_bf, q_bf, vt_bf, att);

  // --- residual + LayerNorm over (S,F) per batch ---
  add_reduce_kernel<<<B_ * 64, 256, 0, stream>>>(x, att, att, partials);
  finalize_stats_kernel<<<B_, 64, 0, stream>>>(partials, stats);
  apply_ln_kernel<<<NT / 256, 256, 0, stream>>>(att, stats, g1, be1, resf, res_bf);

  // --- feed-forward ---
  gemm_ff_kernel<true><<<dim3(64, 16), 128, 0, stream>>>(res_bf, w1_t, b1,
                                                         nullptr, nullptr, h_bf);
  gemm_ff_kernel<false><<<dim3(64, 16), 128, 0, stream>>>(h_bf, w2_t, b2,
                                                          resf, y2, nullptr);

  // --- final LayerNorm ---
  add_reduce_kernel<<<B_ * 64, 256, 0, stream>>>(nullptr, y2, y2, partials);
  finalize_stats_kernel<<<B_, 64, 0, stream>>>(partials, stats);
  apply_ln_kernel<<<NT / 256, 256, 0, stream>>>(y2, stats, g2, be2, out, nullptr);
}
